// IC_14070312861854
// MI455X (gfx1250) — compile-verified, matrix operand order checked by reading, not run on hardware
//
#include <hip/hip_runtime.h>

// -------- CDNA5 (gfx1250, wave32) types --------
typedef __attribute__((ext_vector_type(16))) _Float16 v16h;
typedef __attribute__((ext_vector_type(8)))  float    v8f;
typedef __attribute__((ext_vector_type(4)))  int      i4v;
typedef __attribute__((ext_vector_type(4)))  unsigned int u4v;
typedef __attribute__((ext_vector_type(4)))  float    f4v;

// Expand 16 byte-flags (each 0/1) into 16 f16 lanes-elements (A-matrix fragment).
__device__ __forceinline__ v16h bytes_to_f16(u4v v) {
    v16h r;
    unsigned int w[4] = {v.x, v.y, v.z, v.w};
#pragma unroll
    for (int j = 0; j < 4; ++j)
#pragma unroll
        for (int b = 0; b < 4; ++b)
            r[j * 4 + b] = (_Float16)(float)((w[j] >> (8 * b)) & 0xFFu);
    return r;
}

// Sum 512 values held across a wave32 (16 f16 per lane) with one WMMA:
// D = A(16x32) * ones(32x16); every column of D is the vector of row sums.
// Per-lane sum of the 8 accumulator VGPRs covers M=0..7 (lanes 0-15) or
// M=8..15 (lanes 16-31) of one column; xor-16 shuffle combines the halves.
__device__ __forceinline__ float wmma_wave_sum(v16h vals) {
    v16h ones;
#pragma unroll
    for (int i = 0; i < 16; ++i) ones[i] = (_Float16)1.0f;
    v8f c = {};
    c = __builtin_amdgcn_wmma_f32_16x16x32_f16(
        /*neg_a=*/false, vals, /*neg_b=*/false, ones,
        /*c_mod=*/(short)0, c, /*reuse_a=*/false, /*reuse_b=*/false);
    float p = c[0] + c[1] + c[2] + c[3] + c[4] + c[5] + c[6] + c[7];
    p += __shfl_xor(p, 16);   // all 32 lanes now hold the 512-element total
    return p;
}

// Combine the 8 wave totals of a 256-thread block, one atomicAdd per block.
__device__ __forceinline__ void block_accum(float waveTotal, float* out) {
    __shared__ float wsum[8];
    int lane = threadIdx.x & 31;
    int wave = threadIdx.x >> 5;
    if (lane == 0) wsum[wave] = waveTotal;
    __syncthreads();
    if (threadIdx.x == 0) {
        float s = 0.f;
#pragma unroll
        for (int i = 0; i < 8; ++i) s += wsum[i];
        atomicAdd(out, s);
    }
}

// -------- Kernels --------

// Zero node-state arrays (padded), the log slots in d_out, and the ws dummy.
__global__ void k_init(u4v* __restrict__ active, u4v* __restrict__ newact,
                       u4v* __restrict__ marker, float* __restrict__ logf,
                       float* __restrict__ dummy, int steps) {
    int i = blockIdx.x * blockDim.x + threadIdx.x;   // NPAD/16 threads exactly
    u4v z = {0u, 0u, 0u, 0u};
    active[i] = z; newact[i] = z; marker[i] = z;
    if (i < steps) logf[i] = 0.f;
    if (i == steps) *dummy = 0.f;
}

__global__ void k_seed(const int* __restrict__ seed_idx, int nseeds,
                       unsigned char* __restrict__ active,
                       unsigned char* __restrict__ newact) {
    int i = blockIdx.x * blockDim.x + threadIdx.x;
    if (i < nseeds) {
        int s = seed_idx[i];
        active[s] = 1;   // duplicate seeds collapse, matching .at[].set(1.0)
        newact[s] = 1;
    }
}

// log[0] = sum(active) via WMMA reduction (padded tail is zero).
__global__ void k_count(const u4v* __restrict__ arr, float* __restrict__ cnt) {
    int i = blockIdx.x * blockDim.x + threadIdx.x;   // NPAD/16 threads exactly
    float tot = wmma_wave_sum(bytes_to_f16(arr[i]));
    block_accum(tot, cnt);
}

// Edge pass: frontier gather + Bernoulli test + idempotent scatter of 1.
// rP is single-use -> non-temporal load keeps src/tar/w resident in L2.
__global__ void k_edge(const i4v* __restrict__ src4, const i4v* __restrict__ tar4,
                       const f4v* __restrict__ w4,   const f4v* __restrict__ rp4,
                       const unsigned char* __restrict__ newact,
                       unsigned char* __restrict__ marker, int nvec,
                       const int* __restrict__ src, const int* __restrict__ tar,
                       const float* __restrict__ w, const float* __restrict__ rp,
                       int nEdges) {
    int i = blockIdx.x * blockDim.x + threadIdx.x;
    if (i < nvec) {
        i4v s  = src4[i];
        i4v t  = tar4[i];
        f4v wv = w4[i];
        f4v rv = __builtin_nontemporal_load(&rp4[i]);
        // Cheap local test first (p~0.5) halves the random L2 gathers.
        if (rv.x < wv.x && newact[s.x]) marker[t.x] = 1;
        if (rv.y < wv.y && newact[s.y]) marker[t.y] = 1;
        if (rv.z < wv.z && newact[s.z]) marker[t.z] = 1;
        if (rv.w < wv.w && newact[s.w]) marker[t.w] = 1;
    }
    if (i == 0) {   // scalar tail (empty when E % 4 == 0)
        for (int e = nvec * 4; e < nEdges; ++e)
            if (rp[e] < w[e] && newact[src[e]]) marker[tar[e]] = 1;
    }
}

// Node pass: active|=hit, new = hit & ~old_active, clear marker,
// and accumulate count of updated active (= log of the NEXT step) via WMMA.
__global__ void k_update(u4v* __restrict__ active, u4v* __restrict__ newact,
                         u4v* __restrict__ marker, float* __restrict__ cnt) {
    int i = blockIdx.x * blockDim.x + threadIdx.x;   // NPAD/16 threads exactly
    u4v a  = active[i];
    u4v s  = marker[i];
    u4v na = a | s;          // bytewise: flags are 0/1
    u4v nn = s & ~a;         // newly activated
    active[i] = na;
    newact[i] = nn;
    u4v z = {0u, 0u, 0u, 0u};
    marker[i] = z;
    float tot = wmma_wave_sum(bytes_to_f16(na));
    block_accum(tot, cnt);
}

__global__ void k_final(const unsigned char* __restrict__ act,
                        float* __restrict__ out, int n) {
    int i = blockIdx.x * blockDim.x + threadIdx.x;
    if (i < n) out[i] = (float)act[i];
}

// -------- Host launcher --------
extern "C" void kernel_launch(void* const* d_in, const int* in_sizes, int n_in,
                              void* d_out, int out_size, void* d_ws, size_t ws_size,
                              hipStream_t stream) {
    const int*   src  = (const int*)  d_in[0];
    const int*   tar  = (const int*)  d_in[1];
    const float* w    = (const float*)d_in[2];
    const float* rP   = (const float*)d_in[3];   // [steps, E]
    const int*   seed = (const int*)  d_in[4];

    const int E      = in_sizes[0];
    const int steps  = in_sizes[3] / E;          // 16
    const int nseeds = in_sizes[4];              // 64
    const int N      = out_size - steps;         // 100000

    // Pad node arrays to a multiple of 16B*256thr so count/update kernels
    // launch with full EXEC (WMMA requirement) and no bounds checks.
    const int NPAD       = ((N + 4095) / 4096) * 4096;   // bytes per array
    const int nodeBlocks = NPAD / 4096;                   // 256 thr * 16 B each

    unsigned char* active = (unsigned char*)d_ws;
    unsigned char* newact = active + NPAD;
    unsigned char* marker = newact + NPAD;
    float*         dummy  = (float*)(marker + NPAD);      // 16B-aligned

    float* outAct = (float*)d_out;
    float* logf   = (float*)d_out + N;

    k_init<<<nodeBlocks, 256, 0, stream>>>((u4v*)active, (u4v*)newact,
                                           (u4v*)marker, logf, dummy, steps);
    k_seed<<<(nseeds + 255) / 256, 256, 0, stream>>>(seed, nseeds, active, newact);
    k_count<<<nodeBlocks, 256, 0, stream>>>((const u4v*)active, &logf[0]);

    const int nvec       = E >> 2;
    const int edgeBlocks = (nvec + 255) / 256;

    for (int t = 0; t < steps; ++t) {
        const float* rp = rP + (size_t)t * (size_t)E;
        k_edge<<<edgeBlocks, 256, 0, stream>>>(
            (const i4v*)src, (const i4v*)tar, (const f4v*)w, (const f4v*)rp,
            newact, marker, nvec, src, tar, w, rp, E);
        float* cntOut = (t + 1 < steps) ? &logf[t + 1] : dummy;
        k_update<<<nodeBlocks, 256, 0, stream>>>((u4v*)active, (u4v*)newact,
                                                 (u4v*)marker, cntOut);
    }

    k_final<<<(N + 255) / 256, 256, 0, stream>>>(active, outAct, N);
}
